// Attention_88484916232651
// MI455X (gfx1250) — compile-verified, matrix operand order checked by reading, not run on hardware
//
#include <hip/hip_runtime.h>
#include <hip/hip_bf16.h>

typedef __bf16 bf16_t;
typedef __attribute__((ext_vector_type(16))) __bf16 v16bf;
typedef __attribute__((ext_vector_type(8)))  __bf16 v8bf;
typedef __attribute__((ext_vector_type(8)))  float  v8f;
typedef __attribute__((ext_vector_type(4)))  int    v4i;

#define NEG_INF (-__builtin_inff())

#if __has_builtin(__builtin_amdgcn_global_load_async_to_lds_b128)
#define ASYNC_LDS_OK 1
#endif

// Async DMA one 16-byte chunk global -> LDS (no VGPR round trip, ASYNCcnt).
__device__ inline void async_copy_b128(const bf16_t* g, bf16_t* l) {
#ifdef ASYNC_LDS_OK
    __builtin_amdgcn_global_load_async_to_lds_b128(
        (__attribute__((address_space(1))) v4i*)(unsigned long long)g,
        (__attribute__((address_space(3))) v4i*)(unsigned int)(unsigned long long)l,
        0, 0);
#else
    *(v8bf*)l = *(const v8bf*)g;
#endif
}

__device__ inline void async_join() {
#ifdef ASYNC_LDS_OK
#if __has_builtin(__builtin_amdgcn_s_wait_asynccnt)
    __builtin_amdgcn_s_wait_asynccnt(0);
#else
    asm volatile("s_wait_asynccnt 0x0" ::: "memory");
#endif
#endif
}

// Load a 16-bit WMMA A-style fragment (also serves B when B^T tile is row-major):
// p points at (row_base + khalf); elements [0..7] = K khalf..khalf+7,
// elements [8..15] = K khalf+16..khalf+23 (CDNA5 16-bit A layout).
__device__ inline v16bf load_frag16(const bf16_t* p) {
    union { v16bf v; v8bf h[2]; } u;
    u.h[0] = *(const v8bf*)(p);
    u.h[1] = *(const v8bf*)(p + 16);
    return u.v;
}

// ---------------------------------------------------------------------------
// Kernel 1/4: bf16 GEMM  C[M,N] = A[M,K] * B[N,K]^T   (fp32 accum, bf16 out)
// 256 threads (8 waves). WG tile 128x128; wave tile 64x32 (4x2 wmma).
// Double-buffered LDS, K staged in 64-chunks, async global->LDS DMA.
// ---------------------------------------------------------------------------
__global__ __launch_bounds__(256) void gemm_bf16_nt(
    const bf16_t* __restrict__ A, const bf16_t* __restrict__ B,
    bf16_t* __restrict__ C, int M, int N, int K) {
    __shared__ alignas(16) bf16_t As[2][128 * 64];
    __shared__ alignas(16) bf16_t Bs[2][128 * 64];

    const int tid  = threadIdx.x;
    const int lane = tid & 31;
    const int wave = tid >> 5;
    const int m0 = blockIdx.y * 128;
    const int n0 = blockIdx.x * 128;
    const int wm = (wave & 1) * 64;
    const int wn = (wave >> 1) * 32;

    const int r  = lane & 15;
    const int hi = lane >> 4;
    const int kh = hi * 8;

    // loader coords: each thread moves 64B (4 x b128) per matrix per stage
    const int lrow = tid >> 1;          // 0..127
    const int lcol = (tid & 1) * 32;    // 0 or 32

    v8f acc[4][2] = {};

    auto stage = [&](int buf, int k0) {
        const bf16_t* ag = A + (size_t)(m0 + lrow) * K + k0 + lcol;
        const bf16_t* bg = B + (size_t)(n0 + lrow) * K + k0 + lcol;
        bf16_t* al = As[buf] + lrow * 64 + lcol;
        bf16_t* bl = Bs[buf] + lrow * 64 + lcol;
#ifdef ASYNC_LDS_OK
#pragma unroll
        for (int c = 0; c < 4; ++c) {
            async_copy_b128(ag + c * 8, al + c * 8);
            async_copy_b128(bg + c * 8, bl + c * 8);
        }
#else
        v8bf ta[4], tb[4];
#pragma unroll
        for (int c = 0; c < 4; ++c) { ta[c] = ((const v8bf*)ag)[c]; tb[c] = ((const v8bf*)bg)[c]; }
#pragma unroll
        for (int c = 0; c < 4; ++c) { ((v8bf*)al)[c] = ta[c]; ((v8bf*)bl)[c] = tb[c]; }
#endif
    };

    stage(0, 0);
    async_join();
    __syncthreads();

    const int nk = K >> 6;  // K/64 stages
    for (int kt = 0; kt < nk; ++kt) {
        const int buf = kt & 1;
        if (kt + 1 < nk) stage(buf ^ 1, (kt + 1) * 64);  // overlap DMA with WMMA

        const bf16_t* Asb = As[buf];
        const bf16_t* Bsb = Bs[buf];
#pragma unroll
        for (int kc = 0; kc < 2; ++kc) {
            v16bf a[4], b[2];
#pragma unroll
            for (int mt = 0; mt < 4; ++mt)
                a[mt] = load_frag16(Asb + (wm + mt * 16 + r) * 64 + kc * 32 + kh);
#pragma unroll
            for (int nt = 0; nt < 2; ++nt)
                b[nt] = load_frag16(Bsb + (wn + nt * 16 + r) * 64 + kc * 32 + kh);
#pragma unroll
            for (int mt = 0; mt < 4; ++mt)
#pragma unroll
                for (int nt = 0; nt < 2; ++nt)
                    acc[mt][nt] = __builtin_amdgcn_wmma_f32_16x16x32_bf16(
                        false, a[mt], false, b[nt], (short)0, acc[mt][nt], false, false);
        }
        async_join();      // my DMA for kt+1 landed
        __syncthreads();   // everyone done reading buf / writing buf^1
    }

    // Epilogue. C/D layout: VGPR j = row (j + 8*hi), lane%16 = col.
#pragma unroll
    for (int mt = 0; mt < 4; ++mt)
#pragma unroll
        for (int nt = 0; nt < 2; ++nt) {
            const int col  = n0 + wn + nt * 16 + r;
            const int rowb = m0 + wm + mt * 16 + hi * 8;
            bf16_t* cp = C + (size_t)rowb * N + col;
#pragma unroll
            for (int j = 0; j < 8; ++j)
                cp[(size_t)j * N] = (bf16_t)acc[mt][nt][j];
        }
}

// ---------------------------------------------------------------------------
// Kernel 2/4: interleaved-pair RoPE on q,k slices of qkv buffer (in place).
// qkv row layout: [T, 6144]; heads hh=0..39 get rope (q: 0..31, k: 32..39).
// ---------------------------------------------------------------------------
__global__ __launch_bounds__(256) void rope_kernel(bf16_t* __restrict__ qkv,
                                                   int T, int nheads, int ldq) {
    const int idx = blockIdx.x * blockDim.x + threadIdx.x;
    const int total = T * nheads * 64;
    if (idx >= total) return;
    const int i  = idx & 63;
    const int hh = (idx >> 6) % nheads;
    const int t  = idx / (nheads * 64);
    const float inv_freq = __expf(-(float)i * (9.210340371976184f / 64.0f)); // 10000^(-i/64)
    const float ang = (float)t * inv_freq;
    float s, c;
    __sincosf(ang, &s, &c);
    bf16_t* p = qkv + (size_t)t * ldq + hh * 128 + 2 * i;
    const float x1 = (float)p[0];
    const float x2 = (float)p[1];
    p[0] = (bf16_t)(x1 * c - x2 * s);
    p[1] = (bf16_t)(x1 * s + x2 * c);
}

// ---------------------------------------------------------------------------
// Kernel 3/4: causal GQA flash attention.
// grid = (T/64, H); block = 128 (4 waves); each wave owns 16 query rows.
// qkv: [T,6144] (q: col 0..4095, k: 4096..5119, v: 5120..6143), out: [T,4096].
// ---------------------------------------------------------------------------
__global__ __launch_bounds__(128) void attn_kernel(const bf16_t* __restrict__ qkv,
                                                   bf16_t* __restrict__ out, int T) {
    const int ldq  = 6144;
    const int h    = blockIdx.y;
    const int kvh  = h >> 2;              // H/HKV = 4 heads per group
    const int q_wg = blockIdx.x * 64;
    const int tid  = threadIdx.x;
    const int lane = tid & 31;
    const int wave = tid >> 5;
    const int q0   = q_wg + wave * 16;

    __shared__ alignas(16) bf16_t ks[32 * 128];    // K tile, row-major [key][d]
    __shared__ alignas(16) bf16_t vt[128 * 32];    // V tile transposed [d][key]
    __shared__ alignas(16) bf16_t ps[4][16 * 32];  // per-wave P tile [q][key]

    const int r  = lane & 15;
    const int hi = lane >> 4;
    const int kh = hi * 8;

    // Q fragments for this wave's 16 rows, 4 chunks of K=32 over D=128
    v16bf qa[4];
    {
        const bf16_t* qp = qkv + (size_t)(q0 + r) * ldq + h * 128;
#pragma unroll
        for (int kc = 0; kc < 4; ++kc)
            qa[kc] = load_frag16(qp + kc * 32 + kh);
    }

    float m_i[8], l_i[8];
    v8f   o_acc[8] = {};                 // 8 d-chunks of 16 cols
#pragma unroll
    for (int j = 0; j < 8; ++j) { m_i[j] = NEG_INF; l_i[j] = 0.0f; }

    const float scale = 0.08838834764831845f;  // 1/sqrt(128)
    const int kt_end = (q_wg + 63) >> 5;       // last key tile for this WG

    for (int kt = 0; kt <= kt_end; ++kt) {
        const int kb = kt * 32;
        // --- K tile: async DMA, 4 x b128 per thread ---
        {
            const int krow = tid >> 2;
            const int kcol = (tid & 3) * 32;
            const bf16_t* kp = qkv + (size_t)(kb + krow) * ldq + 4096 + kvh * 128 + kcol;
            bf16_t* kl = ks + krow * 128 + kcol;
#pragma unroll
            for (int c = 0; c < 4; ++c) async_copy_b128(kp + c * 8, kl + c * 8);
        }
        // --- transposed V tile: vt[d][key] (coalesced b16 reads per step) ---
        {
            const int d = tid;  // 0..127
#pragma unroll 4
            for (int key = 0; key < 32; ++key)
                vt[d * 32 + key] = qkv[(size_t)(kb + key) * ldq + 5120 + kvh * 128 + d];
        }
        async_join();
        __syncthreads();

        if (kb <= q0 + 15) {   // causal: this key tile intersects our rows
            // S = Q K^T : two 16-key sub-tiles, K-dim in 4 chunks of 32
            v8f s0 = {}, s1 = {};
#pragma unroll
            for (int kc = 0; kc < 4; ++kc) {
                v16bf b0 = load_frag16(ks + (r)      * 128 + kc * 32 + kh);
                v16bf b1 = load_frag16(ks + (16 + r) * 128 + kc * 32 + kh);
                s0 = __builtin_amdgcn_wmma_f32_16x16x32_bf16(false, qa[kc], false, b0,
                                                             (short)0, s0, false, false);
                s1 = __builtin_amdgcn_wmma_f32_16x16x32_bf16(false, qa[kc], false, b1,
                                                             (short)0, s1, false, false);
            }

            const int key0 = kb + r;
            const int key1 = kb + 16 + r;
            float e0v[8], e1v[8];
#pragma unroll
            for (int j = 0; j < 8; ++j) {
                const int row = q0 + j + 8 * hi;
                float v0 = (key0 <= row) ? s0[j] * scale : NEG_INF;
                float v1 = (key1 <= row) ? s1[j] * scale : NEG_INF;
                float tmax = fmaxf(v0, v1);           // row-max over 32 keys
#pragma unroll
                for (int off = 1; off < 16; off <<= 1)
                    tmax = fmaxf(tmax, __shfl_xor(tmax, off, 32));
                const float mnew  = fmaxf(m_i[j], tmax);
                const float alpha = __expf(m_i[j] - mnew);
                const float e0 = __expf(v0 - mnew);
                const float e1 = __expf(v1 - mnew);
                float psum = e0 + e1;
#pragma unroll
                for (int off = 1; off < 16; off <<= 1)
                    psum += __shfl_xor(psum, off, 32);
                l_i[j] = l_i[j] * alpha + psum;
                m_i[j] = mnew;
#pragma unroll
                for (int dc = 0; dc < 8; ++dc) o_acc[dc][j] *= alpha;
                e0v[j] = e0;
                e1v[j] = e1;
            }

            // stage P (C-layout -> row-major [16q x 32key]) through LDS
            bf16_t* pw = ps[wave];
#pragma unroll
            for (int j = 0; j < 8; ++j) {
                pw[(j + 8 * hi) * 32 + r]      = (bf16_t)e0v[j];
                pw[(j + 8 * hi) * 32 + 16 + r] = (bf16_t)e1v[j];
            }
            // O += P * V : A-frag of P, B-frags from transposed V
            v16bf pa = load_frag16(pw + r * 32 + kh);
#pragma unroll
            for (int dc = 0; dc < 8; ++dc) {
                v16bf vb = load_frag16(vt + (dc * 16 + r) * 32 + kh);
                o_acc[dc] = __builtin_amdgcn_wmma_f32_16x16x32_bf16(
                    false, pa, false, vb, (short)0, o_acc[dc], false, false);
            }
        }
        __syncthreads();
    }

    // normalize and write out [T, H*D]
    float invl[8];
#pragma unroll
    for (int j = 0; j < 8; ++j) invl[j] = 1.0f / l_i[j];
#pragma unroll
    for (int dc = 0; dc < 8; ++dc)
#pragma unroll
        for (int j = 0; j < 8; ++j) {
            const int row = q0 + j + 8 * hi;
            out[(size_t)row * 4096 + h * 128 + dc * 16 + r] =
                (bf16_t)(o_acc[dc][j] * invl[j]);
        }
}

// ---------------------------------------------------------------------------
extern "C" void kernel_launch(void* const* d_in, const int* in_sizes, int n_in,
                              void* d_out, int out_size, void* d_ws, size_t ws_size,
                              hipStream_t stream) {
    const int T = 2048, DIM = 4096, NQKV = 6144;  // (H + 2*HKV) * D
    const bf16_t* x     = (const bf16_t*)d_in[0];
    const bf16_t* w_qkv = (const bf16_t*)d_in[1];
    const bf16_t* w_o   = (const bf16_t*)d_in[2];
    bf16_t* out  = (bf16_t*)d_out;
    bf16_t* qkv  = (bf16_t*)d_ws;                       // [T, 6144]
    bf16_t* attn = qkv + (size_t)T * NQKV;              // [T, 4096]

    // 1) fused QKV projection
    gemm_bf16_nt<<<dim3(NQKV / 128, T / 128), 256, 0, stream>>>(
        x, w_qkv, qkv, T, NQKV, DIM);

    // 2) RoPE on q (32 heads) + k (8 heads) = 40 head-slices
    const int ropeN = T * 40 * 64;
    rope_kernel<<<(ropeN + 255) / 256, 256, 0, stream>>>(qkv, T, 40, NQKV);

    // 3) causal GQA flash attention
    attn_kernel<<<dim3(T / 64, 32), 128, 0, stream>>>(qkv, attn, T);

    // 4) output projection
    gemm_bf16_nt<<<dim3(DIM / 128, T / 128), 256, 0, stream>>>(
        attn, w_o, out, T, DIM, DIM);
}
